// TransDecoder_26362509263558
// MI455X (gfx1250) — matrix operand order repaired and checked by reading, not compiled, not run on hardware
//
#include <hip/hip_runtime.h>
#include <cstdint>
#include <cstddef>

// ---------------------------------------------------------------------------
// MI455X (gfx1250) transformer decoder block.
// All GEMMs + both attention einsums run on v_wmma_f32_16x16x32_f16
// (f16 operands, fp32 accumulate). wave32, 128-thread blocks = 4 waves.
// GEMMs stage tiles in LDS via global_load_async_to_lds_b128 (ASYNCcnt),
// double-buffered so the async engine overlaps the WMMA pipeline.
// ---------------------------------------------------------------------------

typedef _Float16 v16h __attribute__((ext_vector_type(16)));
typedef _Float16 v8h  __attribute__((ext_vector_type(8)));
typedef float    v8f  __attribute__((ext_vector_type(8)));

#define B_ 4
#define N_ 1025
#define L_ 1024
#define E_ 1024
#define H_ 16
#define D_ 64
#define F_ 4096
#define M_ (B_ * N_)      /* 4100 rows on the x side */
#define NPADQ_ 1040       /* 1025 padded to 65 tiles of 16 */

// ---------------- WMMA + async helpers -------------------------------------

__device__ __forceinline__ v8f wmma_f16(v16h a, v16h b, v8f c) {
  return __builtin_amdgcn_wmma_f32_16x16x32_f16(false, a, false, b, (short)0, c,
                                                false, false);
}

// Async global -> LDS 16-byte copy (CDNA5 GLOBAL_LOAD_ASYNC_TO_LDS_B128,
// tracked by ASYNCcnt). lds_off is the LDS byte address (low 32 bits of the
// flat shared-aperture pointer).
__device__ __forceinline__ void gl_async_lds_b128(uint32_t lds_off,
                                                  const void* gaddr) {
  asm volatile("global_load_async_to_lds_b128 %0, %1, off"
               :
               : "v"(lds_off), "v"(gaddr)
               : "memory");
}

__device__ __forceinline__ void wait_async0() {
  asm volatile("s_wait_asynccnt 0x0" ::: "memory");
}

__device__ __forceinline__ uint32_t lds_addr_of(const void* p) {
  return (uint32_t)(uintptr_t)p;  // LDS aperture: low 32 bits == LDS offset
}

// A fragment, 16x32 f16 (documented layout), source row-major, ld halves.
// lane<16: M=lane,    K = {k0..k0+7, k0+16..k0+23}
// lane>=16: M=lane-16, K = {k0+8..k0+15, k0+24..k0+31}
__device__ __forceinline__ v16h load_a16(const _Float16* base, int ld,
                                         int rowbase, int k0, int maxrow,
                                         int hi, int ln15) {
  int r = rowbase + ln15;
  if (r > maxrow) r = maxrow;
  const _Float16* p = base + (size_t)r * ld + k0 + hi * 8;
  v8h lo = *(const v8h*)p;
  v8h hh = *(const v8h*)(p + 16);
  v16h a;
#pragma unroll
  for (int i = 0; i < 8; ++i) { a[i] = lo[i]; a[i + 8] = hh[i]; }
  return a;
}

// B fragment, 32x16 f16: lanes 0-15 hold K={k0..k0+7, k0+16..k0+23}, lanes 16-31 K={+8..+15, +24..+31}.
__device__ __forceinline__ v16h load_b16(const _Float16* base, int ld,
                                         int colbase, int k0, int maxcol,
                                         int hi, int ln15) {
  int c = colbase + ln15;
  if (c > maxcol) c = maxcol;
  const _Float16* p = base + (size_t)c * ld + k0 + hi * 8;
  v8h lo = *(const v8h*)p;
  v8h hh = *(const v8h*)(p + 16);
  v16h b;
#pragma unroll
  for (int i = 0; i < 8; ++i) { b[i] = lo[i]; b[i + 8] = hh[i]; }
  return b;
}

// ---------------- small utility kernels ------------------------------------

__global__ void k_cvt(const float* __restrict__ in, _Float16* __restrict__ out,
                      size_t n) {
  size_t i = (size_t)blockIdx.x * 256 + threadIdx.x;
  if (i < n) out[i] = (_Float16)in[i];
}

__global__ void k_bias3(float* __restrict__ out, const float* __restrict__ p0,
                        const float* __restrict__ p1,
                        const float* __restrict__ p2, int total) {
  int i = blockIdx.x * 256 + threadIdx.x;
  if (i >= total) return;
  int seg = i >> 10, r = i & 1023;
  const float* p = (seg == 0) ? p0 : (seg == 1) ? p1 : p2;
  out[i] = p ? p[r] : 0.f;
}

__global__ __launch_bounds__(256) void k_ln(const float* __restrict__ x,
                                            const float* __restrict__ g,
                                            const float* __restrict__ bt,
                                            _Float16* __restrict__ out) {
  __shared__ float red[256];
  const int row = blockIdx.x, tid = threadIdx.x;
  const float* xr = x + (size_t)row * E_;
  float v[4];
  float s = 0.f;
#pragma unroll
  for (int i = 0; i < 4; ++i) { v[i] = xr[tid + i * 256]; s += v[i]; }
  red[tid] = s;
  __syncthreads();
  for (int st = 128; st > 0; st >>= 1) {
    if (tid < st) red[tid] += red[tid + st];
    __syncthreads();
  }
  float mean = red[0] * (1.0f / E_);
  __syncthreads();
  float s2 = 0.f;
#pragma unroll
  for (int i = 0; i < 4; ++i) { float d0 = v[i] - mean; s2 += d0 * d0; }
  red[tid] = s2;
  __syncthreads();
  for (int st = 128; st > 0; st >>= 1) {
    if (tid < st) red[tid] += red[tid + st];
    __syncthreads();
  }
  float rstd = rsqrtf(red[0] * (1.0f / E_) + 1e-5f);
  _Float16* orow = out + (size_t)row * E_;
#pragma unroll
  for (int i = 0; i < 4; ++i) {
    int c = tid + i * 256;
    orow[c] = (_Float16)((v[i] - mean) * rstd * g[c] + bt[c]);
  }
}

// ---------------- LDS-staged double-buffered WMMA GEMM ---------------------
// out = epi(A @ W^T + bias); A: MxK f16 row-major, W: NoutxK f16 row-major.
// Block = 128 threads = 4 waves; tile = 64 rows x 64 cols, k-step 32.
// Per k-step the block stages A(64x32) and B(64x32) in LDS with async copies;
// each wave computes a 16x64 strip (4 accumulators).
// EPI 0: f16 out = acc+bias
// EPI 1: f32 out = resid + gamma*(acc+bias)
// EPI 2: f16 out = gelu_exact(acc+bias)
template <int EPI>
__global__ __launch_bounds__(128) void k_gemm(
    const _Float16* __restrict__ A, const _Float16* __restrict__ W,
    const float* __restrict__ bias, const float* __restrict__ resid,
    const float* __restrict__ gamma, void* __restrict__ outp, int M, int Nout,
    int K) {
  __shared__ __align__(16) _Float16 Abuf[2][64 * 32];
  __shared__ __align__(16) _Float16 Bbuf[2][64 * 32];

  const int tid = threadIdx.x;
  const int wv = tid >> 5;
  const int lane = tid & 31;
  const int hi = lane >> 4, ln15 = lane & 15;
  const int rowblk = blockIdx.y * 64;
  const int colbase = blockIdx.x * 64;

  // copy assignment: thread copies 16 halves (two b128) of one row of A and B
  const int crow = tid >> 1, cch = tid & 1;  // row 0..63, 16-half chunk 0/1
  int arow = rowblk + crow;
  if (arow > M - 1) arow = M - 1;            // clamp M edge (dup load, safe)
  const _Float16* Ag = A + (size_t)arow * K + cch * 16;
  const _Float16* Bg = W + (size_t)(colbase + crow) * K + cch * 16;
  uint32_t a_lds[2], b_lds[2];
#pragma unroll
  for (int bb = 0; bb < 2; ++bb) {
    a_lds[bb] = lds_addr_of(&Abuf[bb][crow * 32 + cch * 16]);
    b_lds[bb] = lds_addr_of(&Bbuf[bb][crow * 32 + cch * 16]);
  }

  auto issue = [&](int buf, int k0) {
    gl_async_lds_b128(a_lds[buf], Ag + k0);
    gl_async_lds_b128(a_lds[buf] + 16, Ag + k0 + 8);
    gl_async_lds_b128(b_lds[buf], Bg + k0);
    gl_async_lds_b128(b_lds[buf] + 16, Bg + k0 + 8);
  };

  v8f acc[4] = {v8f{}, v8f{}, v8f{}, v8f{}};
  const int KT = K >> 5;
  issue(0, 0);
  for (int kt = 0; kt < KT; ++kt) {
    const int cur = kt & 1;
    wait_async0();        // own async copies for buf[cur] done
    __syncthreads();      // everyone's copies done; prev reads of buf[cur^1] done
    if (kt + 1 < KT) issue(cur ^ 1, (kt + 1) * 32);

    // A fragment for this wave's 16-row strip, from LDS
    const _Float16* ap = &Abuf[cur][(wv * 16 + ln15) * 32 + hi * 8];
    v8h alo = *(const v8h*)ap;
    v8h ahi = *(const v8h*)(ap + 16);
    v16h a;
#pragma unroll
    for (int i = 0; i < 8; ++i) { a[i] = alo[i]; a[i + 8] = ahi[i]; }
#pragma unroll
    for (int j = 0; j < 4; ++j) {
      const _Float16* bp = &Bbuf[cur][(j * 16 + ln15) * 32 + hi * 8];
      v8h blo = *(const v8h*)bp;
      v8h bhi = *(const v8h*)(bp + 16);
      v16h b;
#pragma unroll
      for (int i = 0; i < 8; ++i) { b[i] = blo[i]; b[i + 8] = bhi[i]; }
      acc[j] = wmma_f16(a, b, acc[j]);
    }
    __syncthreads();      // all waves done reading buf[cur] before reuse
  }

#pragma unroll
  for (int j = 0; j < 4; ++j) {
    int col = colbase + j * 16 + ln15;
    float bv = bias ? bias[col] : 0.f;
#pragma unroll
    for (int r = 0; r < 8; ++r) {
      int m = rowblk + wv * 16 + r + hi * 8;
      if (m < M) {
        float v = acc[j][r] + bv;
        size_t off = (size_t)m * Nout + col;
        if (EPI == 0) {
          ((_Float16*)outp)[off] = (_Float16)v;
        } else if (EPI == 1) {
          ((float*)outp)[off] = resid[off] + gamma[col] * v;
        } else {
          float ge = 0.5f * v * (1.0f + erff(v * 0.70710678118654752f));
          ((_Float16*)outp)[off] = (_Float16)ge;
        }
      }
    }
  }
}

// ---------------- head packing ---------------------------------------------

__global__ void k_pack_self(const _Float16* __restrict__ qkv,
                            _Float16* __restrict__ Q, _Float16* __restrict__ K,
                            _Float16* __restrict__ V) {
  size_t idx = (size_t)blockIdx.x * 256 + threadIdx.x;
  const size_t total = (size_t)B_ * H_ * NPADQ_ * D_;
  if (idx >= total) return;
  int d = (int)(idx & 63);
  int n = (int)((idx >> 6) % NPADQ_);
  int h = (int)((idx / ((size_t)64 * NPADQ_)) & 15);
  int bb = (int)(idx / ((size_t)64 * NPADQ_ * 16));
  _Float16 q = (_Float16)0.f, kk = (_Float16)0.f, vv = (_Float16)0.f;
  if (n < N_) {
    const _Float16* row = qkv + ((size_t)bb * N_ + n) * (3 * E_);
    int c = h * 64 + d;
    q = (_Float16)((float)row[c] * 0.125f);  // SCALE = D^-0.5
    kk = row[E_ + c];
    vv = row[2 * E_ + c];
  }
  size_t bh = (size_t)bb * H_ + h;
  Q[(bh * NPADQ_ + n) * 64 + d] = q;
  K[(bh * NPADQ_ + n) * 64 + d] = kk;
  V[(bh * 64 + d) * NPADQ_ + n] = vv;
}

__global__ void k_pack_crossq(const _Float16* __restrict__ qo,
                              _Float16* __restrict__ Q) {
  size_t idx = (size_t)blockIdx.x * 256 + threadIdx.x;
  const size_t total = (size_t)B_ * H_ * NPADQ_ * D_;
  if (idx >= total) return;
  int d = (int)(idx & 63);
  int n = (int)((idx >> 6) % NPADQ_);
  int h = (int)((idx / ((size_t)64 * NPADQ_)) & 15);
  int bb = (int)(idx / ((size_t)64 * NPADQ_ * 16));
  _Float16 q = (_Float16)0.f;
  if (n < N_)
    q = (_Float16)((float)qo[((size_t)bb * N_ + n) * E_ + h * 64 + d] * 0.125f);
  Q[(((size_t)bb * H_ + h) * NPADQ_ + n) * 64 + d] = q;
}

__global__ void k_pack_crosskv(const _Float16* __restrict__ kvo,
                               _Float16* __restrict__ K,
                               _Float16* __restrict__ V) {
  size_t idx = (size_t)blockIdx.x * 256 + threadIdx.x;
  const size_t total = (size_t)B_ * H_ * L_ * D_;
  if (idx >= total) return;
  int d = (int)(idx & 63);
  int n = (int)((idx >> 6) & 1023);
  int h = (int)((idx >> 16) & 15);
  int bb = (int)(idx >> 20);
  const _Float16* row = kvo + ((size_t)bb * L_ + n) * (2 * E_);
  int c = h * 64 + d;
  size_t bh = (size_t)bb * H_ + h;
  K[(bh * (size_t)L_ + n) * 64 + d] = row[c];
  V[(bh * 64 + d) * (size_t)L_ + n] = row[E_ + c];
}

// ---------------- fused attention ------------------------------------------
// One block (4 waves) per (qtile16, h, b).  Phase 1: S = Q@K^T + bias (WMMA,
// scores to LDS f32).  Phase 2: softmax in LDS, P f16 to LDS.  Phase 3:
// O = P @ V^T (WMMA, A fragments re-read from LDS).

__device__ __forceinline__ int rel_idx(int p, int q) {
  int dh = (p >> 5) - (q >> 5) + 31;
  int dw = (p & 31) - (q & 31) + 31;
  return dh * 63 + dw;  // (2*Ww-1) = 63
}

template <int SELF>
__global__ __launch_bounds__(128) void k_attn(const _Float16* __restrict__ Q,
                                              const _Float16* __restrict__ Km,
                                              const _Float16* __restrict__ Vt,
                                              const float* __restrict__ rpb,
                                              _Float16* __restrict__ outp) {
  constexpr int NPAD = SELF ? NPADQ_ : L_;
  constexpr int KVLEN = SELF ? N_ : L_;
  constexpr int NT = NPAD / 16;

  extern __shared__ char smem_raw[];
  float* S = (float*)smem_raw;                          // [16][NPAD] f32
  _Float16* P = (_Float16*)(smem_raw + 16 * NPAD * 4);  // [16][NPAD] f16
  float* red = (float*)(smem_raw + 16 * NPAD * 6);      // [16][8]
  float* rowred = red + 128;                            // [16]

  const int tid = threadIdx.x, wv = tid >> 5, lane = tid & 31;
  const int hi = lane >> 4, ln15 = lane & 15;
  const int qbase = blockIdx.x * 16;
  const int h = blockIdx.y, bb = blockIdx.z;
  const size_t bh = (size_t)bb * H_ + h;
  const _Float16* Qp = Q + bh * (size_t)NPADQ_ * 64;
  const _Float16* Kp = Km + bh * (size_t)NPAD * 64;
  const _Float16* Vp = Vt + bh * 64 * (size_t)NPAD;

  // ---- Phase 1: scores ----
  v16h aq0 = load_a16(Qp, 64, qbase, 0, NPADQ_ - 1, hi, ln15);
  v16h aq1 = load_a16(Qp, 64, qbase, 32, NPADQ_ - 1, hi, ln15);
  for (int t = wv; t < NT; t += 4) {
    v16h b0 = load_b16(Kp, 64, t * 16, 0, KVLEN - 1, hi, ln15);
    v16h b1 = load_b16(Kp, 64, t * 16, 32, KVLEN - 1, hi, ln15);
    v8f acc = v8f{};
    acc = wmma_f16(aq0, b0, acc);
    acc = wmma_f16(aq1, b1, acc);
    int j = t * 16 + ln15;
#pragma unroll
    for (int r = 0; r < 8; ++r) {
      int m = r + hi * 8, i = qbase + m;
      float sc = acc[r];
      if (i < N_ && j < KVLEN) {
        int idx;
        if (SELF) {
          if (i == 0 && j == 0)      idx = 3971;   // nrd-1
          else if (i == 0)           idx = 3969;   // nrd-3
          else if (j == 0)           idx = 3970;   // nrd-2
          else                       idx = rel_idx(i - 1, j - 1);
          if (j > i) sc += -1e9f;                  // causal mask
        } else {
          idx = (i == 0) ? 3969 : rel_idx(i - 1, j);
        }
        sc += rpb[idx * H_ + h];
      }
      S[m * NPAD + j] = sc;
    }
  }
  __syncthreads();

  // ---- Phase 2: softmax ----
  {
    int row = tid >> 3, slot = tid & 7;
    float* Sr = S + row * NPAD;
    float mx = -3.0e38f;
    for (int j = slot; j < KVLEN; j += 8) mx = fmaxf(mx, Sr[j]);
    red[row * 8 + slot] = mx;
    __syncthreads();
    if (slot == 0) {
      float m2 = red[row * 8];
#pragma unroll
      for (int t2 = 1; t2 < 8; ++t2) m2 = fmaxf(m2, red[row * 8 + t2]);
      rowred[row] = m2;
    }
    __syncthreads();
    float rm = rowred[row];
    float sum = 0.f;
    for (int j = slot; j < KVLEN; j += 8) {
      float e = __expf(Sr[j] - rm);
      Sr[j] = e;
      sum += e;
    }
    red[row * 8 + slot] = sum;
    __syncthreads();
    if (slot == 0) {
      float s2 = 0.f;
#pragma unroll
      for (int t2 = 0; t2 < 8; ++t2) s2 += red[row * 8 + t2];
      rowred[row] = s2;
    }
    __syncthreads();
    float inv = 1.0f / rowred[row];
    _Float16* Pr = P + row * NPAD;
    for (int j = slot; j < NPAD; j += 8)
      Pr[j] = (_Float16)((j < KVLEN) ? Sr[j] * inv : 0.f);
  }
  __syncthreads();

  // ---- Phase 3: O = P @ V^T (each wave owns one 16-wide d tile) ----
  const int dt = wv;
  v8f acc = v8f{};
  for (int c = 0; c < NPAD / 32; ++c) {
    const _Float16* pp = P + ln15 * NPAD + c * 32 + hi * 8;
    v8h lo = *(const v8h*)pp;
    v8h hh = *(const v8h*)(pp + 16);
    v16h a;
#pragma unroll
    for (int i2 = 0; i2 < 8; ++i2) { a[i2] = lo[i2]; a[i2 + 8] = hh[i2]; }
    v16h b = load_b16(Vp, NPAD, dt * 16, c * 32, 63, hi, ln15);
    acc = wmma_f16(a, b, acc);
  }
  int d = dt * 16 + ln15;
#pragma unroll
  for (int r = 0; r < 8; ++r) {
    int i = qbase + r + hi * 8;
    if (i < N_)
      outp[((size_t)bb * N_ + i) * E_ + h * 64 + d] = (_Float16)acc[r];
  }
}

// ---------------- host orchestration ---------------------------------------

extern "C" void kernel_launch(void* const* d_in, const int* in_sizes, int n_in,
                              void* d_out, int out_size, void* d_ws,
                              size_t ws_size, hipStream_t stream) {
  (void)in_sizes; (void)n_in; (void)out_size; (void)ws_size;
  const float* x         = (const float*)d_in[0];
  const float* y         = (const float*)d_in[1];
  const float* n1g       = (const float*)d_in[2];
  const float* n1b       = (const float*)d_in[3];
  const float* n2g       = (const float*)d_in[4];
  const float* n2b       = (const float*)d_in[5];
  const float* n3g       = (const float*)d_in[6];
  const float* n3b       = (const float*)d_in[7];
  const float* sa_qkv_w  = (const float*)d_in[8];
  const float* sa_q_bias = (const float*)d_in[9];
  const float* sa_v_bias = (const float*)d_in[10];
  const float* sa_rpb    = (const float*)d_in[11];
  const float* sa_proj_w = (const float*)d_in[12];
  const float* sa_proj_b = (const float*)d_in[13];
  const float* ca_q_w    = (const float*)d_in[14];
  const float* ca_kv_w   = (const float*)d_in[15];
  const float* ca_q_bias = (const float*)d_in[16];
  const float* ca_v_bias = (const float*)d_in[17];
  const float* ca_rpb    = (const float*)d_in[18];
  const float* ca_proj_w = (const float*)d_in[19];
  const float* ca_proj_b = (const float*)d_in[20];
  const float* fc1_w     = (const float*)d_in[21];
  const float* fc1_b     = (const float*)d_in[22];
  const float* fc2_w     = (const float*)d_in[23];
  const float* fc2_b     = (const float*)d_in[24];
  const float* gamma1    = (const float*)d_in[25];
  const float* gamma2    = (const float*)d_in[26];
  const float* gamma3    = (const float*)d_in[27];
  float* outx = (float*)d_out;

  char* ws = (char*)d_ws;
  size_t off = 0;
  auto take = [&](size_t bytes) -> char* {
    char* p = ws + off;
    off += (bytes + 255) & ~(size_t)255;
    return p;
  };

  _Float16* w_qkv    = (_Float16*)take((size_t)3 * E_ * E_ * 2);
  _Float16* w_saproj = (_Float16*)take((size_t)E_ * E_ * 2);
  _Float16* w_caq    = (_Float16*)take((size_t)E_ * E_ * 2);
  _Float16* w_cakv   = (_Float16*)take((size_t)2 * E_ * E_ * 2);
  _Float16* w_caproj = (_Float16*)take((size_t)E_ * E_ * 2);
  _Float16* w_fc1    = (_Float16*)take((size_t)F_ * E_ * 2);
  _Float16* w_fc2    = (_Float16*)take((size_t)E_ * F_ * 2);
  float*    qkv_bias = (float*)take(3 * E_ * 4);
  float*    kv_bias  = (float*)take(2 * E_ * 4);
  _Float16* yh       = (_Float16*)take((size_t)B_ * L_ * E_ * 2);
  _Float16* hln      = (_Float16*)take((size_t)M_ * E_ * 2);
  // BIG1 is time-shared: qkv out -> cross-kv out -> cross-q out -> fc1 out
  char*     BIG1     = take((size_t)M_ * F_ * 2);
  _Float16* qs       = (_Float16*)take((size_t)B_ * H_ * NPADQ_ * D_ * 2);
  _Float16* ks       = (_Float16*)take((size_t)B_ * H_ * NPADQ_ * D_ * 2);
  _Float16* vts      = (_Float16*)take((size_t)B_ * H_ * D_ * NPADQ_ * 2);
  _Float16* attno    = (_Float16*)take((size_t)M_ * E_ * 2);
  float*    x1       = (float*)take((size_t)M_ * E_ * 4);
  float*    x2       = (float*)take((size_t)M_ * E_ * 4);
  _Float16* kc       = (_Float16*)take((size_t)B_ * H_ * L_ * D_ * 2);
  _Float16* vct      = (_Float16*)take((size_t)B_ * H_ * D_ * L_ * 2);
  _Float16* qc       = qs;  // self-attn done before cross-q pack

  auto cvt = [&](const float* src, _Float16* dst, size_t n) {
    k_cvt<<<(unsigned)((n + 255) / 256), 256, 0, stream>>>(src, dst, n);
  };

  cvt(sa_qkv_w, w_qkv, (size_t)3 * E_ * E_);
  cvt(sa_proj_w, w_saproj, (size_t)E_ * E_);
  cvt(ca_q_w, w_caq, (size_t)E_ * E_);
  cvt(ca_kv_w, w_cakv, (size_t)2 * E_ * E_);
  cvt(ca_proj_w, w_caproj, (size_t)E_ * E_);
  cvt(fc1_w, w_fc1, (size_t)F_ * E_);
  cvt(fc2_w, w_fc2, (size_t)E_ * F_);
  cvt(y, yh, (size_t)B_ * L_ * E_);
  k_bias3<<<(3 * E_ + 255) / 256, 256, 0, stream>>>(qkv_bias, sa_q_bias,
                                                    nullptr, sa_v_bias, 3 * E_);
  k_bias3<<<(2 * E_ + 255) / 256, 256, 0, stream>>>(kv_bias, nullptr, ca_v_bias,
                                                    nullptr, 2 * E_);

  const unsigned packN = (unsigned)(((size_t)B_ * H_ * NPADQ_ * D_ + 255) / 256);
  const unsigned packL = (unsigned)(((size_t)B_ * H_ * L_ * D_ + 255) / 256);
  const dim3 blk(128);
  const dim3 attn_grid(NPADQ_ / 16, H_, B_);
  const size_t smem_self = 16 * NPADQ_ * 6 + (128 + 16) * 4;  // 100,416 B
  const size_t smem_cross = 16 * L_ * 6 + (128 + 16) * 4;     //  98,880 B

  // -- self attention branch --
  k_ln<<<M_, 256, 0, stream>>>(x, n1g, n1b, hln);
  k_gemm<0><<<dim3(3 * E_ / 64, (M_ + 63) / 64), blk, 0, stream>>>(
      hln, w_qkv, qkv_bias, nullptr, nullptr, BIG1, M_, 3 * E_, E_);
  k_pack_self<<<packN, 256, 0, stream>>>((const _Float16*)BIG1, qs, ks, vts);
  k_attn<1><<<attn_grid, blk, smem_self, stream>>>(qs, ks, vts, sa_rpb, attno);
  k_gemm<1><<<dim3(E_ / 64, (M_ + 63) / 64), blk, 0, stream>>>(
      attno, w_saproj, sa_proj_b, x, gamma1, x1, M_, E_, E_);

  // -- cross attention branch --
  k_gemm<0><<<dim3(2 * E_ / 64, (B_ * L_) / 64), blk, 0, stream>>>(
      yh, w_cakv, kv_bias, nullptr, nullptr, BIG1, B_ * L_, 2 * E_, E_);
  k_pack_crosskv<<<packL, 256, 0, stream>>>((const _Float16*)BIG1, kc, vct);
  k_ln<<<M_, 256, 0, stream>>>(x1, n2g, n2b, hln);
  k_gemm<0><<<dim3(E_ / 64, (M_ + 63) / 64), blk, 0, stream>>>(
      hln, w_caq, ca_q_bias, nullptr, nullptr, BIG1, M_, E_, E_);
  k_pack_crossq<<<packN, 256, 0, stream>>>((const _Float16*)BIG1, qc);
  k_attn<0><<<attn_grid, blk, smem_cross, stream>>>(qc, kc, vct, ca_rpb, attno);
  k_gemm<1><<<dim3(E_ / 64, (M_ + 63) / 64), blk, 0, stream>>>(
      attno, w_caproj, ca_proj_b, x1, gamma2, x2, M_, E_, E_);

  // -- FFN --
  k_ln<<<M_, 256, 0, stream>>>(x2, n3g, n3b, hln);
  k_gemm<2><<<dim3(F_ / 64, (M_ + 63) / 64), blk, 0, stream>>>(
      hln, w_fc1, fc1_b, nullptr, nullptr, BIG1, M_, F_, E_);
  k_gemm<1><<<dim3(E_ / 64, (M_ + 63) / 64), blk, 0, stream>>>(
      (const _Float16*)BIG1, w_fc2, fc2_b, x2, gamma3, outx, M_, E_, F_);
}
